// GLBernVAEModule_32512902431537
// MI455X (gfx1250) — matrix operand order repaired and checked by reading, not compiled, not run on hardware
//
#include <hip/hip_runtime.h>
#include <hip/hip_bf16.h>

typedef __attribute__((ext_vector_type(16))) _Float16 v16h;
typedef __attribute__((ext_vector_type(8)))  float    v8f;

// ---- constants derived from the reference ----
#define BB   4
#define NN   800
#define TT   16
#define HH   16
#define BN   3200           // B*N rows
#define COLS 3200           // columns of fc1 GEMM per channel i
#define NTILE 200           // COLS/16
#define JT    50            // N/16
#define MAIN_TILES 40000    // N * JT
#define MAIN_BLOCKS 5000    // MAIN_TILES / 8 waves
#define BN_COUNT 51200.0f   // B*N*H elements per channel
#define LOSS_DENOM 81920000.0f // B*N*N*2H
#define TEMP_INV 5.0f
#define SLOPE 0.01f

static __device__ __forceinline__ v8f wmma16(v16h a, v16h b, v8f c) {
  return __builtin_amdgcn_wmma_f32_16x16x32_f16(
      /*neg_a=*/false, a, /*neg_b=*/false, b,
      /*c_mod=*/(short)0, c, /*reuse_a=*/false, /*reuse_b=*/false);
}

// K index for 16-bit A-matrix element e of a lane (hl = lane>>4)
static __device__ __forceinline__ int a_kidx(int hl, int e) {
  return hl * 16 + e;
}

// ---------------- Kernel 1: node encoder h0 = lrelu(x @ W_enc^T + b_enc) ----------------
// D(16xH x 16 rows) = W_enc(16x16, K-padded to 32) x x^T.  200 tiles, one wave each.
__global__ void k_encode(const float* __restrict__ x, const float* __restrict__ Wenc,
                         const float* __restrict__ benc,
                         float* __restrict__ h0f, _Float16* __restrict__ h0h) {
  const int gwave = (blockIdx.x * blockDim.x + threadIdx.x) >> 5;
  if (gwave >= NTILE) return;
  const int lane = threadIdx.x & 31;
  const int nn = lane & 15;
  const int hl = lane >> 4;

  v16h a;
#pragma unroll
  for (int e = 0; e < 16; ++e) {
    const int K = a_kidx(hl, e);
    a[e] = (K < 16) ? (_Float16)Wenc[nn * 16 + K] : (_Float16)0.0f;
  }
  const int row = gwave * 16 + nn;   // h0 row handled by this lane (both halves)
  v16h b;
#pragma unroll
  for (int e = 0; e < 16; ++e) b[e] = (_Float16)0.0f;
  if (hl == 0) {
    const float* xr = x + row * TT;
#pragma unroll
    for (int e = 0; e < 16; ++e) b[e] = (_Float16)xr[e];
  }
  v8f c = {};
  c = wmma16(a, b, c);
  // element r -> output channel M = hl*8 + r, column = row
  float* df = h0f + row * HH + hl * 8;
  _Float16* dh = h0h + row * HH + hl * 8;
#pragma unroll
  for (int r = 0; r < 8; ++r) {
    float v = c[r] + benc[hl * 8 + r];
    v = (v > 0.0f) ? v : SLOPE * v;
    df[r] = v;
    dh[r] = (_Float16)v;
  }
}

// ---------------- Kernel 2: fc1 + per-channel BN stats + q = Wb.h_raw ----------------
// One block per channel i. Each wave: 25 tiles of 16 (b,j) columns, one WMMA per tile.
__global__ void k_fc1_stats(const _Float16* __restrict__ h0h,
                            const float* __restrict__ W1, const float* __restrict__ b1,
                            const float* __restrict__ Wb,
                            float* __restrict__ q, float* __restrict__ stats) {
  const int i = blockIdx.x;
  const int lane = threadIdx.x & 31;
  const int wave = threadIdx.x >> 5;   // 0..7
  const int nn = lane & 15;
  const int hl = lane >> 4;

  v16h a;
#pragma unroll
  for (int e = 0; e < 16; ++e) a[e] = (_Float16)W1[nn * 32 + a_kidx(hl, e)];

  float bsub[8], wsub[8];
#pragma unroll
  for (int r = 0; r < 8; ++r) {
    bsub[r] = b1[hl * 8 + r];
    wsub[r] = Wb[hl * 8 + r];
  }

  float s1 = 0.0f, s2 = 0.0f;
  for (int t = wave; t < NTILE; t += 8) {
    const int c0 = t * 16;                 // column base; N%16==0 -> no b straddle
    const int bidx = c0 / NN;
    const int j = c0 % NN + nn;
    const int rowj = bidx * NN + j;
    const int rowi = bidx * NN + i;
    const int row = hl ? rowj : rowi;      // K 0..15 -> h0_i, K 16..31 -> h0_j
    v16h b = *(const v16h*)(h0h + row * HH);
    v8f acc = {};
    acc = wmma16(a, b, acc);
    float qp = 0.0f;
#pragma unroll
    for (int r = 0; r < 8; ++r) {
      float v = fmaxf(acc[r] + bsub[r], 0.0f);   // relu
      s1 += v;
      s2 += v * v;
      qp += wsub[r] * v;
    }
    float qo = qp + __shfl_xor(qp, 16);
    if (hl == 0) q[i * COLS + c0 + nn] = qo;
  }
#pragma unroll
  for (int off = 16; off > 0; off >>= 1) {
    s1 += __shfl_xor(s1, off);
    s2 += __shfl_xor(s2, off);
  }
  __shared__ float ls1[8], ls2[8];
  if (lane == 0) { ls1[wave] = s1; ls2[wave] = s2; }
  __syncthreads();
  if (threadIdx.x == 0) {
    float t1 = 0.0f, t2 = 0.0f;
    for (int w = 0; w < 8; ++w) { t1 += ls1[w]; t2 += ls2[w]; }
    stats[2 * i] = t1;
    stats[2 * i + 1] = t2;
  }
}

// ---------------- Kernel 3: fold BN into per-channel affine (alpha, beta) ----------------
__global__ void k_bn_final(const float* __restrict__ stats,
                           const float* __restrict__ bn_w, const float* __restrict__ bn_b,
                           const float* __restrict__ Wb, const float* __restrict__ bb,
                           float* __restrict__ ab) {
  const int i = blockIdx.x * blockDim.x + threadIdx.x;
  if (i >= NN) return;
  const float mean = stats[2 * i] / BN_COUNT;
  const float var  = stats[2 * i + 1] / BN_COUNT - mean * mean;
  const float inv  = rsqrtf(var + 1e-5f);
  const float alpha = inv * bn_w[i];
  float sumWb = 0.0f;
#pragma unroll
  for (int h = 0; h < HH; ++h) sumWb += Wb[h];
  const float beta = (bn_b[i] - mean * alpha) * sumWb + bb[0];
  ab[2 * i] = alpha;
  ab[2 * i + 1] = beta;
}

// ---------------- Kernel 4: sample z, decoder WMMA, adj + loss partials ----------------
// 5000 blocks x 8 waves; wave handles one (i, 16-j tile) for all b.
__global__ void k_main(const float* __restrict__ h0f,
                       const float* __restrict__ q, const float* __restrict__ ab,
                       const float* __restrict__ u,
                       const float* __restrict__ W2, const float* __restrict__ b2,
                       const float* __restrict__ W3, const float* __restrict__ b3,
                       float* __restrict__ adj, float* __restrict__ lossPart) {
  const int lane = threadIdx.x & 31;
  const int wave = threadIdx.x >> 5;
  const int tile = blockIdx.x * 8 + wave;   // 0..39999
  const int i = tile / JT;
  const int j0 = (tile % JT) * 16;
  const int nn = lane & 15;
  const int hl = lane >> 4;
  const int j = j0 + nn;

  // A matrices: W3 top(rows 0..15) and bottom(rows 16..31), K padded 16->32
  v16h aT, aB;
#pragma unroll
  for (int e = 0; e < 16; ++e) {
    const int K = a_kidx(hl, e);
    aT[e] = (K < 16) ? (_Float16)W3[nn * HH + K] : (_Float16)0.0f;
    aB[e] = (K < 16) ? (_Float16)W3[(nn + 16) * HH + K] : (_Float16)0.0f;
  }
  v8f cT, cB;
#pragma unroll
  for (int r = 0; r < 8; ++r) {
    cT[r] = b3[hl * 8 + r];
    cB[r] = b3[16 + hl * 8 + r];
  }
  const float alpha = ab[2 * i];
  const float beta  = ab[2 * i + 1];
  float w2l[16], b2l[16];
#pragma unroll
  for (int h = 0; h < HH; ++h) { w2l[h] = W2[h]; b2l[h] = b2[h]; }

  float zsum = 0.0f, loss = 0.0f;
  for (int bi = 0; bi < BB; ++bi) {
    v16h bm;
#pragma unroll
    for (int e = 0; e < 16; ++e) bm[e] = (_Float16)0.0f;
    if (hl == 0) {
      const float s  = alpha * q[i * COLS + bi * NN + j] + beta;
      const float uu = u[bi * NN * NN + i * NN + j];
      const float noise = logf(uu) - log1pf(-uu);
      const float z = 1.0f / (1.0f + expf(-(s + noise) * TEMP_INV));
      zsum += z;
#pragma unroll
      for (int h = 0; h < HH; ++h) {
        float hv = z * w2l[h] + b2l[h];
        hv = (hv > 0.0f) ? hv : SLOPE * hv;   // lrelu
        bm[h] = (_Float16)hv;
      }
    }
    v8f dT = wmma16(aT, bm, cT);
    v8f dB = wmma16(aB, bm, cB);
    const float* xpi = h0f + (bi * NN + i) * HH + hl * 8;
    const float* xpj = h0f + (bi * NN + j) * HH + hl * 8;
#pragma unroll
    for (int r = 0; r < 8; ++r) {
      const float d1 = dT[r] - xpi[r];
      const float d2 = dB[r] - xpj[r];
      loss += d1 * d1 + d2 * d2;
    }
  }
  if (hl == 0) adj[i * NN + j] = zsum * 0.25f;

#pragma unroll
  for (int off = 16; off > 0; off >>= 1) loss += __shfl_xor(loss, off);
  __shared__ float lp[8];
  if (lane == 0) lp[wave] = loss;
  __syncthreads();
  if (threadIdx.x == 0) {
    float t = 0.0f;
    for (int w = 0; w < 8; ++w) t += lp[w];
    lossPart[blockIdx.x] = t;
  }
}

// ---------------- Kernel 5: fixed-order loss reduction ----------------
__global__ void k_loss_reduce(const float* __restrict__ lossPart, float* __restrict__ out) {
  __shared__ float sm[256];
  float s = 0.0f;
  for (int idx = threadIdx.x; idx < MAIN_BLOCKS; idx += 256) s += lossPart[idx];
  sm[threadIdx.x] = s;
  __syncthreads();
  for (int off = 128; off > 0; off >>= 1) {
    if ((int)threadIdx.x < off) sm[threadIdx.x] += sm[threadIdx.x + off];
    __syncthreads();
  }
  if (threadIdx.x == 0) out[NN * NN] = sm[0] / LOSS_DENOM;
}

extern "C" void kernel_launch(void* const* d_in, const int* in_sizes, int n_in,
                              void* d_out, int out_size, void* d_ws, size_t ws_size,
                              hipStream_t stream) {
  (void)in_sizes; (void)n_in; (void)out_size; (void)ws_size;
  const float* x     = (const float*)d_in[0];
  const float* u     = (const float*)d_in[1];
  const float* W_enc = (const float*)d_in[2];
  const float* b_enc = (const float*)d_in[3];
  const float* W1    = (const float*)d_in[4];
  const float* b1    = (const float*)d_in[5];
  const float* Wb    = (const float*)d_in[6];
  const float* bbias = (const float*)d_in[7];
  // d_in[8..11] = Wmu,bmu,Wlv,blv : dead code, never read
  const float* bn_w  = (const float*)d_in[12];
  const float* bn_b  = (const float*)d_in[13];
  const float* W2    = (const float*)d_in[14];
  const float* b2    = (const float*)d_in[15];
  const float* W3    = (const float*)d_in[16];
  const float* b3    = (const float*)d_in[17];

  float* out = (float*)d_out;  // [0,640000) adj, [640000] loss

  // workspace layout (floats)
  float*     ws   = (float*)d_ws;
  float*     h0f  = ws;                              // 51200
  _Float16*  h0h  = (_Float16*)(ws + 51200);         // 51200 halfs (25600 slots)
  float*     q    = ws + 76800;                      // 2,560,000
  float*     stats= ws + 2636800;                    // 1600
  float*     abuf = ws + 2638400;                    // 1600
  float*     lpar = ws + 2640000;                    // 5000

  k_encode<<<25, 256, 0, stream>>>(x, W_enc, b_enc, h0f, h0h);
  k_fc1_stats<<<NN, 256, 0, stream>>>(h0h, W1, b1, Wb, q, stats);
  k_bn_final<<<4, 256, 0, stream>>>(stats, bn_w, bn_b, Wb, bbias, abuf);
  k_main<<<MAIN_BLOCKS, 256, 0, stream>>>(h0f, q, abuf, u, W2, b2, W3, b3, out, lpar);
  k_loss_reduce<<<1, 256, 0, stream>>>(lpar, out);
}
